// VectorQuantizer_82506321756728
// MI455X (gfx1250) — matrix operand order repaired and checked by reading, not compiled, not run on hardware
//
#include <hip/hip_runtime.h>
#include <hip/hip_bf16.h>

typedef __attribute__((ext_vector_type(16))) _Float16 v16h;
typedef __attribute__((ext_vector_type(8)))  float    v8f;

union V16H { v16h h; uint4 u[2]; };

#define N_ROWS 32768
#define DIM 256
#define K_EMB 8192
#define KT 32               // codebook rows per LDS stage
#define LDS_STRIDE 528      // 256 f16 (512B) + 16B pad -> conflict-free banks
#define NSTAGES (K_EMB / KT)

// CDNA5 async global->LDS copy, 16B per lane (tracked by ASYNCcnt)
__device__ __forceinline__ void async_copy_b128(unsigned ldsOff, const void* gaddr) {
    asm volatile("global_load_async_to_lds_b128 %0, %1, off"
                 :: "v"(ldsOff), "v"(gaddr) : "memory");
}
__device__ __forceinline__ void wait_async0() {
    asm volatile("s_wait_asynccnt 0x0" ::: "memory");
}

// ---------------- prep: zh = (f16)(-2 * z) ----------------
__global__ void vq_prep_z(const float* __restrict__ z, _Float16* __restrict__ zh) {
    int i = (blockIdx.x * blockDim.x + threadIdx.x) * 4;
    float4 v = *(const float4*)(z + i);
    union { _Float16 h[4]; uint2 u; } p;
    p.h[0] = (_Float16)(-2.0f * v.x);
    p.h[1] = (_Float16)(-2.0f * v.y);
    p.h[2] = (_Float16)(-2.0f * v.z);
    p.h[3] = (_Float16)(-2.0f * v.w);
    *(uint2*)(zh + i) = p.u;
}

// ---------------- prep: eh = (f16)E, norms[k] = ||E_k||^2 (f32 exact) ----------------
__global__ void vq_prep_e(const float* __restrict__ E, _Float16* __restrict__ eh,
                          float* __restrict__ norms) {
    int wave = threadIdx.x >> 5;
    int lane = threadIdx.x & 31;
    int row  = blockIdx.x * 8 + wave;
    const float* src = E + (long)row * DIM + lane * 8;
    union { _Float16 h[8]; uint4 u; } p;
    float s = 0.f;
#pragma unroll
    for (int t = 0; t < 8; ++t) {
        float v = src[t];
        s += v * v;
        p.h[t] = (_Float16)v;
    }
    *(uint4*)(eh + (long)row * DIM + lane * 8) = p.u;
#pragma unroll
    for (int m = 1; m < 32; m <<= 1) s += __shfl_xor(s, m, 32);
    if (lane == 0) norms[row] = s;
}

// ---------------- main: fused GEMM + argmin + gather ----------------
__global__ void __launch_bounds__(256) vq_main(
    const _Float16* __restrict__ zh, const _Float16* __restrict__ eh,
    const float* __restrict__ norms, const float* __restrict__ Ef32,
    float* __restrict__ out)
{
    __shared__ __align__(16) unsigned char smem[2][KT * LDS_STRIDE];

    const int tid  = threadIdx.x;
    const int lane = tid & 31;
    const int wave = tid >> 5;
    const int half = lane >> 4;     // which 16-lane half of the wave
    const int n16  = lane & 15;     // column / row-in-tile index
    const int row0 = (blockIdx.x * 8 + wave) * 32;   // 2 x 16-row tiles per wave

    // LDS byte offsets of the two stage buffers (low 32 bits of flat LDS address)
    const unsigned ldsBase0 = (unsigned)(size_t)&smem[0][0];
    const unsigned ldsBase1 = (unsigned)(size_t)&smem[1][0];

    // Two resident A tiles: 32x256 f16 in 128 VGPRs; each B tile feeds both.
    V16H a[2][8];
#pragma unroll
    for (int t = 0; t < 2; ++t) {
        const _Float16* zr = zh + (long)(row0 + t * 16 + n16) * DIM + half * 8;
#pragma unroll
        for (int s = 0; s < 8; ++s) {
            a[t][s].u[0] = *(const uint4*)(zr + 32 * s);
            a[t][s].u[1] = *(const uint4*)(zr + 32 * s + 16);
        }
    }

    float bestV[2][8];
    int   bestI[2][8];
#pragma unroll
    for (int t = 0; t < 2; ++t)
#pragma unroll
        for (int j = 0; j < 8; ++j) { bestV[t][j] = 3.4e38f; bestI[t][j] = 0; }

    // stage-copy coordinates (4 x 16B chunks per thread per stage)
    unsigned co[4];     // LDS offset within a buffer
    long     go[4];     // f16-element offset within eh for row base
#pragma unroll
    for (int i = 0; i < 4; ++i) {
        int cid = tid + 256 * i;
        int r = cid >> 5;
        int c = cid & 31;
        co[i] = (unsigned)(r * LDS_STRIDE + c * 16);
        go[i] = (long)r * DIM + c * 8;
    }

    // prologue: async-stage 0 -> buffer 0
#pragma unroll
    for (int i = 0; i < 4; ++i)
        async_copy_b128(ldsBase0 + co[i], eh + go[i]);
    wait_async0();
    __syncthreads();

    int cur = 0;
    for (int stage = 0; stage < NSTAGES; ++stage) {
        const int k0 = stage * KT;

        // async-prefetch next stage into the other buffer; DMA overlaps the 32 WMMAs
        if (stage + 1 < NSTAGES) {
            const unsigned nbase = (cur ? ldsBase0 : ldsBase1);
            const _Float16* gsrc = eh + (long)(k0 + KT) * DIM;
#pragma unroll
            for (int i = 0; i < 4; ++i)
                async_copy_b128(nbase + co[i], gsrc + go[i]);
        }

        const unsigned char* buf = &smem[cur][0];
#pragma unroll
        for (int kk = 0; kk < KT; kk += 16) {
            const int kIdx = k0 + kk + n16;
            const float nv = norms[kIdx];           // off critical path: added post-WMMA

            // B tile: lane = column n16; half h takes K=[8h..8h+7] and [16+8h..16+8h+7] of each slab
            const unsigned char* brow = buf + (kk + n16) * LDS_STRIDE + half * 16;
            V16H b[8];
#pragma unroll
            for (int s = 0; s < 8; ++s) {
                b[s].u[0] = *(const uint4*)(brow + s * 64);
                b[s].u[1] = *(const uint4*)(brow + s * 64 + 32);
            }

            v8f c0 = {};   // inline-0 C operand; norm folded in after
            v8f c1 = {};
#pragma unroll
            for (int s = 0; s < 8; ++s) {
                c0 = __builtin_amdgcn_wmma_f32_16x16x32_f16(
                        false, a[0][s].h, false, b[s].h, (short)0, c0, false, false);
                c1 = __builtin_amdgcn_wmma_f32_16x16x32_f16(
                        false, a[1][s].h, false, b[s].h, (short)0, c1, false, false);
            }

            // dist = ||e_k||^2 - 2 z_m . e_k
#pragma unroll
            for (int j = 0; j < 8; ++j) {
                float d0 = c0[j] + nv;
                if (d0 < bestV[0][j]) { bestV[0][j] = d0; bestI[0][j] = kIdx; }
                float d1 = c1[j] + nv;
                if (d1 < bestV[1][j]) { bestV[1][j] = d1; bestI[1][j] = kIdx; }
            }
        }

        // publish next buffer: own async writes done, then block-wide barrier
        wait_async0();
        __syncthreads();
        cur ^= 1;
    }

    // cross-lane argmin within each 16-lane half (first-index tie-break)
#pragma unroll
    for (int t = 0; t < 2; ++t) {
#pragma unroll
        for (int j = 0; j < 8; ++j) {
            float v = bestV[t][j];
            int   i = bestI[t][j];
#pragma unroll
            for (int mm = 1; mm < 16; mm <<= 1) {
                float vo = __shfl_xor(v, mm, 16);
                int   io = __shfl_xor(i, mm, 16);
                if (vo < v || (vo == v && io < i)) { v = vo; i = io; }
            }
            bestI[t][j] = i;
        }
    }

    // gather winning f32 embedding rows -> output (coalesced 16-lane float4 copies)
    const float4* E4 = (const float4*)Ef32;
    float4*       O4 = (float4*)out;
#pragma unroll
    for (int t = 0; t < 2; ++t) {
#pragma unroll
        for (int j = 0; j < 8; ++j) {
            long row = row0 + t * 16 + half * 8 + j;
            long idx = bestI[t][j];
#pragma unroll
            for (int u = 0; u < 4; ++u)
                O4[row * 64 + n16 + 16 * u] = E4[idx * 64 + n16 + 16 * u];
        }
    }
}

extern "C" void kernel_launch(void* const* d_in, const int* in_sizes, int n_in,
                              void* d_out, int out_size, void* d_ws, size_t ws_size,
                              hipStream_t stream) {
    const float* z = (const float*)d_in[0];        // [8,4096,256] f32
    const float* E = (const float*)d_in[1];        // [8192,256]  f32
    float* out = (float*)d_out;                    // [8,4096,256] f32

    char* ws = (char*)d_ws;
    _Float16* zh    = (_Float16*)ws;                       // 16 MB
    _Float16* eh    = (_Float16*)(ws + 16777216);          //  4 MB
    float*    norms = (float*)   (ws + 20971520);          // 32 KB

    vq_prep_z<<<(N_ROWS * DIM) / (256 * 4), 256, 0, stream>>>(z, zh);
    vq_prep_e<<<K_EMB / 8, 256, 0, stream>>>(E, eh, norms);
    vq_main<<<N_ROWS / 256, 256, 0, stream>>>(zh, eh, norms, E, out);
}
